// MultiHeadAttention_50783693308389
// MI455X (gfx1250) — matrix operand order repaired and checked by reading, not compiled, not run on hardware
//
#include <hip/hip_runtime.h>

// ---------------------------------------------------------------------------
// MI455X (gfx1250, wave32) fused attention-without-softmax:
//   context = (x Wq^T + bq) @ [ (x Wk^T)^T (x Wv^T) ] * 1/sqrt(D)
// 5 bf16 WMMA GEMMs (f32 accumulate), re-associated to avoid the NxN
// intermediate. All GEMMs: C = A * B^T, A[MxK] / B[NxK] row-major, K-contig.
// Global->LDS staging uses gfx1250 async-to-LDS ops (ASYNCcnt).
// ---------------------------------------------------------------------------

typedef __bf16 bf16;
typedef __attribute__((ext_vector_type(16))) __bf16 v16bf;
typedef __attribute__((ext_vector_type(8)))  __bf16 v8bf;
typedef __attribute__((ext_vector_type(4)))  __bf16 bf16x4;
typedef __attribute__((ext_vector_type(8)))  float  v8f;
typedef __attribute__((ext_vector_type(4)))  float  f32x4;

#define TILE_M 128
#define TILE_N 256
#define TILE_K 32
#define LDSK   (TILE_K + 8)   // padded row (80B) to spread LDS banks

union FragU  { v16bf v; f32x4 q[2]; };
union PackU  { v8bf  v; f32x4 q;    };

// gfx1250 async global->LDS copy, 16B per lane, tracked by ASYNCcnt.
__device__ __forceinline__ void async_copy_b128(unsigned lds_off, const void* gptr) {
    asm volatile("global_load_async_to_lds_b128 %0, %1, off"
                 :: "v"(lds_off), "v"((unsigned long long)(uintptr_t)gptr)
                 : "memory");
}
__device__ __forceinline__ unsigned lds_addr(const void* p) {
    return (unsigned)(uintptr_t)p;   // LDS aperture: addr[31:0] is the LDS offset
}

// ---------------------------------------------------------------------------
// fp32 -> bf16 conversion (4 elements / thread)
// ---------------------------------------------------------------------------
__global__ __launch_bounds__(256)
void cvt_f32_to_bf16(const float* __restrict__ in, bf16* __restrict__ out, int n) {
    int i = (blockIdx.x * 256 + threadIdx.x) * 4;
    if (i < n) {
        f32x4 f = *(const f32x4*)(in + i);
        bf16x4 o;
        o[0] = (bf16)f[0];
        o[1] = (bf16)f[1];
        o[2] = (bf16)f[2];
        o[3] = (bf16)f[3];
        *(bf16x4*)(out + i) = o;
    }
}

// ---------------------------------------------------------------------------
// Tiled WMMA GEMM: C = alpha * (A * B^T) + bias
//   A: [M x K] bf16 row-major;  B: [N x K] bf16 row-major
//   C: OUT_BF16 ? bf16 : f32 ;  TRANS_OUT stores C^T ([N x M], ld = M)
// 256 threads = 8 waves (2x4); block tile 128x256; wave tile 64x64
// (4x4 WMMA tiles -> 16 v_wmma per K-step, 4x fragment reuse).
// ---------------------------------------------------------------------------
template <bool TRANS_OUT, bool OUT_BF16>
__global__ __launch_bounds__(256)
void wmma_gemm_bf16(const bf16* __restrict__ A,
                    const bf16* __restrict__ B,
                    void* __restrict__ Cout,
                    const float* __restrict__ bias,
                    int M, int N, int K, float alpha) {
    __shared__ bf16 sA[2][TILE_M * LDSK];   // 20 KB
    __shared__ bf16 sB[2][TILE_N * LDSK];   // 40 KB

    const int tid = threadIdx.x;
    const int bm  = blockIdx.y * TILE_M;
    const int bn  = blockIdx.x * TILE_N;

    // ---- async staging ownership:
    //   A: thread t -> row t>>1, 32B chunk at col (t&1)*16   (2 x b128)
    //   B: thread t -> row t,    full 64B row                 (4 x b128)
    const int arow = tid >> 1;
    const int acol = (tid & 1) * 16;
    const bf16* Aload = A + (size_t)(bm + arow) * K + acol;
    const bf16* Bload = B + (size_t)(bn + tid) * K;

    unsigned sAoff[2], sBoff[2];
    sAoff[0] = lds_addr(&sA[0][arow * LDSK + acol]);
    sAoff[1] = lds_addr(&sA[1][arow * LDSK + acol]);
    sBoff[0] = lds_addr(&sB[0][tid * LDSK]);
    sBoff[1] = lds_addr(&sB[1][tid * LDSK]);

    // ---- wave tiling: 2 (M) x 4 (N) waves, each 64x64
    const int w    = tid >> 5;
    const int lane = tid & 31;
    const int m0   = (w >> 2) * 64;
    const int n0   = (w & 3) * 64;
    const int lr   = lane & 15;   // row/col within a 16x16 tile
    const int hi   = lane >> 4;   // lane-half select

    v8f acc[4][4] = {};

    const int S = K / TILE_K;

    // issue K-step 0 into buffer 0 (6 async b128 per thread)
    {
        async_copy_b128(sAoff[0],      Aload);
        async_copy_b128(sAoff[0] + 16, Aload + 8);
        async_copy_b128(sBoff[0],      Bload);
        async_copy_b128(sBoff[0] + 16, Bload + 8);
        async_copy_b128(sBoff[0] + 32, Bload + 16);
        async_copy_b128(sBoff[0] + 48, Bload + 24);
    }

    for (int s = 0; s < S; ++s) {
        const int  cur  = s & 1;
        const bool more = (s + 1 < S);

        if (more) {   // issue next K-step into the other buffer
            const int nb = cur ^ 1;
            const bf16* ap = Aload + (size_t)(s + 1) * TILE_K;
            const bf16* bp = Bload + (size_t)(s + 1) * TILE_K;
            async_copy_b128(sAoff[nb],      ap);
            async_copy_b128(sAoff[nb] + 16, ap + 8);
            async_copy_b128(sBoff[nb],      bp);
            async_copy_b128(sBoff[nb] + 16, bp + 8);
            async_copy_b128(sBoff[nb] + 32, bp + 16);
            async_copy_b128(sBoff[nb] + 48, bp + 24);
            if (s + 2 < S) {   // warm L2 for the tile after next
                __builtin_prefetch(Aload + (size_t)(s + 2) * TILE_K, 0, 1);
                __builtin_prefetch(Bload + (size_t)(s + 2) * TILE_K, 0, 1);
            }
            asm volatile("s_wait_asynccnt 0x6" ::: "memory");  // current buf done
        } else {
            asm volatile("s_wait_asynccnt 0x0" ::: "memory");
        }
        __syncthreads();   // all waves' async data visible

        const bf16* sAb = &sA[cur][0];
        const bf16* sBb = &sB[cur][0];

        // A fragments: lane holds row m; K = hi*8..+7 and hi*8+16..+23
        v16bf afr[4];
#pragma unroll
        for (int tm = 0; tm < 4; ++tm) {
            const bf16* p = sAb + (m0 + tm * 16 + lr) * LDSK + hi * 8;
            FragU u;
            u.q[0] = *(const f32x4*)(p);
            u.q[1] = *(const f32x4*)(p + 16);
            afr[tm] = u.v;
        }
        // B fragments: lane holds col n; K = hi*8..+7 and hi*8+16..+23
        v16bf bfr[4];
#pragma unroll
        for (int tn = 0; tn < 4; ++tn) {
            const bf16* p = sBb + (n0 + tn * 16 + lr) * LDSK + hi * 8;
            FragU u;
            u.q[0] = *(const f32x4*)(p);
            u.q[1] = *(const f32x4*)(p + 16);
            bfr[tn] = u.v;
        }

#pragma unroll
        for (int tm = 0; tm < 4; ++tm)
#pragma unroll
            for (int tn = 0; tn < 4; ++tn)
                acc[tm][tn] = __builtin_amdgcn_wmma_f32_16x16x32_bf16(
                    false, afr[tm], false, bfr[tn],
                    (short)0, acc[tm][tn], false, false);

        __syncthreads();   // protect buf[cur] before iter s+1 re-issues into it
    }

    // ---- epilogue: C/D layout per 16x16 tile: VGPR r -> M = r + hi*8, N = lr
#pragma unroll
    for (int tm = 0; tm < 4; ++tm) {
#pragma unroll
        for (int tn = 0; tn < 4; ++tn) {
            const int nIdx = bn + n0 + tn * 16 + lr;
            const float bv = bias ? bias[nIdx] : 0.0f;
            const int mBase = bm + m0 + tm * 16 + hi * 8;
            if (TRANS_OUT && OUT_BF16) {
                // consecutive r -> consecutive addresses in C^T: one b128 store
                PackU u;
#pragma unroll
                for (int r = 0; r < 8; ++r)
                    u.v[r] = (bf16)(acc[tm][tn][r] * alpha + bv);
                *(f32x4*)((bf16*)Cout + (size_t)nIdx * M + mBase) = u.q;
            } else {
#pragma unroll
                for (int r = 0; r < 8; ++r) {
                    const float val = acc[tm][tn][r] * alpha + bv;
                    if (OUT_BF16)
                        ((bf16*)Cout)[(size_t)(mBase + r) * N + nIdx] = (bf16)val;
                    else
                        ((float*)Cout)[(size_t)(mBase + r) * N + nIdx] = val;
                }
            }
        }
    }
}

// ---------------------------------------------------------------------------
// Host-side orchestration
// ---------------------------------------------------------------------------
extern "C" void kernel_launch(void* const* d_in, const int* in_sizes, int n_in,
                              void* d_out, int out_size, void* d_ws, size_t ws_size,
                              hipStream_t stream) {
    constexpr int N = 4096;
    constexpr int D = 2048;
    const float scale = 0.022097086912079608f;  // 1/sqrt(2048)

    const float* x  = (const float*)d_in[0];
    const float* Wq = (const float*)d_in[1];
    const float* bq = (const float*)d_in[2];
    const float* Wk = (const float*)d_in[3];
    const float* bk = (const float*)d_in[4];
    const float* Wv = (const float*)d_in[5];
    const float* bv = (const float*)d_in[6];
    float* out = (float*)d_out;

    // workspace carve-up (bf16 elements): 48M elems = 96 MB (L2-resident)
    bf16* xb  = (bf16*)d_ws;
    bf16* Wqb = xb  + (size_t)N * D;
    bf16* Wkb = Wqb + (size_t)D * D;
    bf16* Wvb = Wkb + (size_t)D * D;
    bf16* Qb  = Wvb + (size_t)D * D;           // [N x D]
    bf16* Ktb = Qb  + (size_t)N * D;           // [D x N] = K^T
    bf16* Vtb = Ktb + (size_t)N * D;           // [D x N] = V^T
    bf16* Sb  = Vtb + (size_t)N * D;           // [D x D] = (V^T K) * scale

    dim3 blk(256);

    // 1) fp32 -> bf16 conversions
    cvt_f32_to_bf16<<<(N * D) / 1024, blk, 0, stream>>>(x,  xb,  N * D);
    cvt_f32_to_bf16<<<(D * D) / 1024, blk, 0, stream>>>(Wq, Wqb, D * D);
    cvt_f32_to_bf16<<<(D * D) / 1024, blk, 0, stream>>>(Wk, Wkb, D * D);
    cvt_f32_to_bf16<<<(D * D) / 1024, blk, 0, stream>>>(Wv, Wvb, D * D);

    // 2) QKV projections (C = x W^T + b); K and V stored transposed
    dim3 gProj(D / TILE_N, N / TILE_M);
    wmma_gemm_bf16<false, true><<<gProj, blk, 0, stream>>>(xb, Wqb, Qb,  bq, N, D, D, 1.0f);
    wmma_gemm_bf16<true,  true><<<gProj, blk, 0, stream>>>(xb, Wkb, Ktb, bk, N, D, D, 1.0f);
    wmma_gemm_bf16<true,  true><<<gProj, blk, 0, stream>>>(xb, Wvb, Vtb, bv, N, D, D, 1.0f);

    // 3) S' = (V^T K) * scale  -> [D x D] bf16 ;  S'[c2][c1] = Vt·Kt^T
    dim3 gS(D / TILE_N, D / TILE_M);
    wmma_gemm_bf16<false, true><<<gS, blk, 0, stream>>>(Vtb, Ktb, Sb, nullptr, D, D, N, scale);

    // 4) context = Q · S'^T  -> f32 output [N x D]
    dim3 gCtx(D / TILE_N, N / TILE_M);
    wmma_gemm_bf16<false, false><<<gCtx, blk, 0, stream>>>(Qb, Sb, out, nullptr, N, D, D, 1.0f);
}
